// EMSNET_45105746542626
// MI455X (gfx1250) — matrix operand order repaired and checked by reading, not compiled, not run on hardware
//
#include <hip/hip_runtime.h>
#include <stdint.h>

// ---- CDNA5 WMMA fragment types -------------------------------------------
typedef __attribute__((ext_vector_type(2)))  float  v2f;
typedef __attribute__((ext_vector_type(8)))  float  v8f;
typedef __attribute__((ext_vector_type(16))) __bf16 v16bf;

union BF16Frag { unsigned int u[8]; uint4 q[2]; v16bf v; };

#define LRELU(x) ((x) >= 0.0f ? (x) : 0.1f * (x))

__device__ __forceinline__ unsigned short f2bf(float f) {
    union { float f; unsigned u; } c; c.f = f;
    unsigned u = c.u;
    unsigned r = (u + 0x7FFFu + ((u >> 16) & 1u)) >> 16;   // RNE
    return (unsigned short)r;
}

// ---- CDNA5 async global->LDS copy (ASYNCcnt path), with safe fallback ----
// Probe round 2: param0 is '__vector_size__(4*sizeof(int)) int' in addrspace(1)
// => signature (as1 v4i* src, as3 v4i* dst, imm offset, imm cpol).
#if __has_builtin(__builtin_amdgcn_global_load_async_to_lds_b128)
#define USE_ASYNC_LDS 1
typedef int av4i __attribute__((__vector_size__(4 * sizeof(int))));
typedef __attribute__((address_space(1))) av4i ga_v4i;
typedef __attribute__((address_space(3))) av4i la_v4i;
#else
#define USE_ASYNC_LDS 0
#endif

__device__ __forceinline__ void async_copy16(const unsigned short* g, unsigned short* l) {
#if USE_ASYNC_LDS
    __builtin_amdgcn_global_load_async_to_lds_b128((ga_v4i*)g, (la_v4i*)l, 0, 0);
#else
    *(uint4*)l = *(const uint4*)g;     // synchronous staged fallback
#endif
}

template <int N>
__device__ __forceinline__ void wait_async() {
#if USE_ASYNC_LDS
#if __has_builtin(__builtin_amdgcn_s_wait_asynccnt)
    __builtin_amdgcn_s_wait_asynccnt(N);
#else
    asm volatile("s_wait_asynccnt %0" :: "i"(N) : "memory");
#endif
#endif
}

// ===========================================================================
// K1: local branch. One wave (32 lanes) per (b,c) sequence; activations live
// in LDS ping-pong buffers (1152 floats/seq). Weights stream from global and
// stay resident in WGP$/L2 (only ~96 KB total); prefetch the hot ones.
// Output: feat[140800][64] = time-max after lc6.
// ===========================================================================
__global__ void local_branch_kernel(const float* __restrict__ x,
    const float* __restrict__ lc1_w, const float* __restrict__ lc1_b,
    const float* __restrict__ lc2_w, const float* __restrict__ lc2_b,
    const float* __restrict__ lc3_w, const float* __restrict__ lc3_b,
    const float* __restrict__ lc4_w, const float* __restrict__ lc4_b,
    const float* __restrict__ lc5_w, const float* __restrict__ lc5_b,
    const float* __restrict__ lc6_w, const float* __restrict__ lc6_b,
    float* __restrict__ feat)
{
    __shared__ float bufA[8][576];
    __shared__ float bufB[8][576];
    const int w    = threadIdx.x >> 5;
    const int lane = threadIdx.x & 31;
    const int seq  = blockIdx.x * 8 + w;     // 17600 blocks * 8 = 140800

    if (threadIdx.x < 8) {                   // keep weights warm (global_prefetch_b8)
        __builtin_prefetch(lc2_w + threadIdx.x * 160, 0, 3);
        __builtin_prefetch(lc4_w + threadIdx.x * 384, 0, 3);
        __builtin_prefetch(lc5_w + threadIdx.x * 768, 0, 3);
        __builtin_prefetch(lc6_w + threadIdx.x * 1536, 0, 3);
    }

    float* A  = bufA[w];
    float* Bb = bufB[w];

    for (int i = lane; i < 40; i += 32) Bb[i] = x[seq * 40 + i];
    __syncthreads();

    // lc1: 1->16 ch, k=5, len 40->36
    for (int idx = lane; idx < 576; idx += 32) {
        int o = idx / 36, p = idx % 36;
        float s = lc1_b[o];
#pragma unroll
        for (int k = 0; k < 5; ++k) s += Bb[p + k] * lc1_w[o * 5 + k];
        A[idx] = LRELU(s);
    }
    __syncthreads();

    // lc2: 16->16, k=5, len 36->32, maxpool2 -> 16x16
    for (int idx = lane; idx < 256; idx += 32) {
        int o = idx / 16, q = idx % 16;
        float m = -3.4e38f;
#pragma unroll
        for (int p2 = 0; p2 < 2; ++p2) {
            int p = 2 * q + p2;
            float s = lc2_b[o];
            for (int ci = 0; ci < 16; ++ci) {
                const float* wr = lc2_w + (o * 16 + ci) * 5;
                const float* ar = A + ci * 36 + p;
#pragma unroll
                for (int k = 0; k < 5; ++k) s += ar[k] * wr[k];
            }
            m = fmaxf(m, LRELU(s));
        }
        Bb[o * 16 + q] = m;
    }
    __syncthreads();

    // lc3: 16->32, k=3, len 16->14
    for (int idx = lane; idx < 448; idx += 32) {
        int o = idx / 14, p = idx % 14;
        float s = lc3_b[o];
        for (int ci = 0; ci < 16; ++ci) {
            const float* wr = lc3_w + (o * 16 + ci) * 3;
            const float* ar = Bb + ci * 16 + p;
            s += ar[0] * wr[0] + ar[1] * wr[1] + ar[2] * wr[2];
        }
        A[idx] = LRELU(s);
    }
    __syncthreads();

    // lc4: 32->32, k=3, len 14->12, maxpool2 -> 32x6
    for (int idx = lane; idx < 192; idx += 32) {
        int o = idx / 6, q = idx % 6;
        float m = -3.4e38f;
#pragma unroll
        for (int p2 = 0; p2 < 2; ++p2) {
            int p = 2 * q + p2;
            float s = lc4_b[o];
            for (int ci = 0; ci < 32; ++ci) {
                const float* wr = lc4_w + (o * 32 + ci) * 3;
                const float* ar = A + ci * 14 + p;
                s += ar[0] * wr[0] + ar[1] * wr[1] + ar[2] * wr[2];
            }
            m = fmaxf(m, LRELU(s));
        }
        Bb[o * 6 + q] = m;
    }
    __syncthreads();

    // lc5: 32->64, k=3, len 6->4
    for (int idx = lane; idx < 256; idx += 32) {
        int o = idx / 4, p = idx % 4;
        float s = lc5_b[o];
        for (int ci = 0; ci < 32; ++ci) {
            const float* wr = lc5_w + (o * 32 + ci) * 3;
            const float* ar = Bb + ci * 6 + p;
            s += ar[0] * wr[0] + ar[1] * wr[1] + ar[2] * wr[2];
        }
        A[idx] = LRELU(s);
    }
    __syncthreads();

    // lc6: 64->64, k=3, len 4->2, then max over time
#pragma unroll
    for (int h = 0; h < 2; ++h) {
        int o = lane + 32 * h;
        float m = -3.4e38f;
#pragma unroll
        for (int p = 0; p < 2; ++p) {
            float s = lc6_b[o];
            for (int ci = 0; ci < 64; ++ci) {
                const float* wr = lc6_w + (o * 64 + ci) * 3;
                const float* ar = A + ci * 4 + p;
                s += ar[0] * wr[0] + ar[1] * wr[1] + ar[2] * wr[2];
            }
            m = fmaxf(m, LRELU(s));
        }
        feat[seq * 64 + o] = m;
    }
}

// ===========================================================================
// K2: ld dense as fp32 WMMA GEMM: (140800x64) @ ld_w^T (64x64).
// V_WMMA_F32_16X16X4_F32; epilogue fuses bias + lrelu + w1 and scatters bf16
// into the c1 input matrix U[512][17664] (cols 0..17599).
// ===========================================================================
__global__ void ld_gemm_kernel(const float* __restrict__ feat,
                               const float* __restrict__ ld_w,
                               const float* __restrict__ ld_b,
                               const float* __restrict__ w1p,
                               unsigned short* __restrict__ U)
{
    const int wid  = (blockIdx.x * blockDim.x + threadIdx.x) >> 5;
    const int lane = threadIdx.x & 31;
    const int tM = wid >> 2;            // 8800 row tiles
    const int tN = wid & 3;             // 4 col tiles
    const int half = lane >> 4;
    const int l16  = lane & 15;
    const int m = tM * 16 + l16;
    const int n = tN * 16 + l16;

    const float* arow = feat + (size_t)m * 64;
    const float* brow = ld_w + (size_t)n * 64;   // B[k][n] = ld_w[n][k]
    v8f acc = {};
#pragma unroll 4
    for (int kk = 0; kk < 64; kk += 4) {
        v2f a, b;
        a.x = arow[kk + 2 * half + 0];   // A 16x4: K = 2*half, 2*half+1
        a.y = arow[kk + 2 * half + 1];
        b.x = brow[kk + 2 * half + 0];   // B 4x16 mirrors A K-split
        b.y = brow[kk + 2 * half + 1];
        acc = __builtin_amdgcn_wmma_f32_16x16x4_f32(false, a, false, b,
                                                    (short)0, acc, false, false);
    }
    const float w1   = w1p[0];
    const float bias = ld_b[n];
#pragma unroll
    for (int i = 0; i < 8; ++i) {
        int row = tM * 16 + i + 8 * half;        // C/D: M = vgpr + 8*(lane>=16)
        float v = LRELU(acc[i] + bias) * w1;
        int b_ = row / 275, c_ = row % 275;      // reshape (B*C,64)->(B,C*64)
        U[(size_t)b_ * 17664 + c_ * 64 + n] = f2bf(v);
    }
}

// ===========================================================================
// K3: global branch. One 256-thread block per batch. x[b] (44 KB) staged in
// LDS; gc1 (9x40) -> lrelu -> gc2 1x1 -> lrelu -> spatial max -> gd dense ->
// lrelu -> bf16 * w2 into U cols 17600..17663.
// ===========================================================================
__global__ void global_branch_kernel(const float* __restrict__ x,
    const float* __restrict__ gc1_w, const float* __restrict__ gc1_b,
    const float* __restrict__ gc2_w, const float* __restrict__ gc2_b,
    const float* __restrict__ gd_w,  const float* __restrict__ gd_b,
    const float* __restrict__ w2p,
    unsigned short* __restrict__ U)
{
    __shared__ float xs[11000];          // 275*40
    __shared__ float a1s[16 * 267];
    __shared__ float pm[4 * 64];
    __shared__ float gvec[64];
    const int b   = blockIdx.x;
    const int tid = threadIdx.x;

    for (int i = tid; i < 11000; i += 256) xs[i] = x[(size_t)b * 11000 + i];
    __syncthreads();

    for (int idx = tid; idx < 16 * 267; idx += 256) {
        int o = idx / 267, i = idx % 267;
        float s = gc1_b[o];
        for (int dy = 0; dy < 9; ++dy) {
            const float* xr = xs + (i + dy) * 40;
            const float* wr = gc1_w + o * 360 + dy * 40;
#pragma unroll 8
            for (int dx = 0; dx < 40; ++dx) s += xr[dx] * wr[dx];
        }
        a1s[o * 267 + i] = LRELU(s);
    }
    __syncthreads();

    {   // gc2 1x1 + running max over 267 positions (4 chunks x 64 channels)
        int o2 = tid & 63, chunk = tid >> 6;
        float m = -3.4e38f;
        for (int i = chunk; i < 267; i += 4) {
            float s = gc2_b[o2];
#pragma unroll
            for (int ci = 0; ci < 16; ++ci) s += a1s[ci * 267 + i] * gc2_w[o2 * 16 + ci];
            m = fmaxf(m, LRELU(s));
        }
        pm[chunk * 64 + o2] = m;
    }
    __syncthreads();
    if (tid < 64)
        gvec[tid] = fmaxf(fmaxf(pm[tid], pm[64 + tid]),
                          fmaxf(pm[128 + tid], pm[192 + tid]));
    __syncthreads();
    if (tid < 64) {
        float s = gd_b[tid];
        for (int k = 0; k < 64; ++k) s += gvec[k] * gd_w[tid * 64 + k];
        U[(size_t)b * 17664 + 17600 + tid] = f2bf(w2p[0] * LRELU(s));
    }
}

// ===========================================================================
// K4: fp32 -> bf16 weight conversion (c1_w).
// ===========================================================================
__global__ void f32_to_bf16_kernel(const float* __restrict__ in,
                                   unsigned short* __restrict__ out, int n)
{
    int i = blockIdx.x * blockDim.x + threadIdx.x;
    if (i < n) out[i] = f2bf(in[i]);
}

// ===========================================================================
// K5: c1 GEMM, bf16 WMMA 16x16x32, LDS-tiled with double-buffered async
// global->LDS staging (ASYNCcnt). M=512, N=512, K=17664 (552 chunks of 32).
// Block 256 thr = 8 waves; block tile 64x64; each wave one 16x32 strip
// (A-fragment reused across 2 WMMAs). Fragments are exact 16B runs in the
// CDNA5 bf16 A/B layouts -> ds_load_b128 x2 per fragment.
// ===========================================================================
#define C1_K     17664
#define C1_STEPS 552

__global__ void __launch_bounds__(256)
c1_gemm_bf16_kernel(const unsigned short* __restrict__ U,
                    const unsigned short* __restrict__ Wb,
                    const float* __restrict__ bias,
                    float* __restrict__ out)
{
    __shared__ unsigned short As[2][64 * 32];
    __shared__ unsigned short Bs[2][64 * 32];

    const int tid  = threadIdx.x;
    const int wv   = tid >> 5;
    const int lane = tid & 31;
    const int m0 = blockIdx.x * 64;
    const int n0 = blockIdx.y * 64;
    const int mw = (wv >> 1) * 16;           // wave row strip within block
    const int nw = (wv & 1) * 32;            // wave col strip within block
    const int half = lane >> 4;
    const int l16  = lane & 15;

    // staging: each thread owns one 16B chunk of A and one of B per k-chunk
    const int srow = tid >> 2;               // 0..63
    const int skg  = (tid & 3) * 8;          // element offset {0,8,16,24}
    const unsigned short* gA = U  + (size_t)(m0 + srow) * C1_K + skg;
    const unsigned short* gB = Wb + (size_t)(n0 + srow) * C1_K + skg;
    const int lofs = srow * 32 + skg;

    v8f acc0 = {}, acc1 = {};

    async_copy16(gA, &As[0][lofs]);          // prologue: chunk 0 -> buffer 0
    async_copy16(gB, &Bs[0][lofs]);

    for (int c = 0; c < C1_STEPS; ++c) {
        const int cur = c & 1;
        if (c + 1 < C1_STEPS) {              // prefetch next chunk -> other buf
            const size_t go = (size_t)(c + 1) * 32;
            async_copy16(gA + go, &As[cur ^ 1][lofs]);
            async_copy16(gB + go, &Bs[cur ^ 1][lofs]);
            wait_async<2>();                 // in-order: current chunk landed
        } else {
            wait_async<0>();
        }
        __syncthreads();                     // whole tile visible to all waves

        BF16Frag av, bv0, bv1;
        const unsigned short* ar  = &As[cur][(mw + l16) * 32];
        av.q[0]  = *(const uint4*)(ar + 8 * half);        // A: K 8h..8h+7
        av.q[1]  = *(const uint4*)(ar + 16 + 8 * half);   //    K 16+8h..+7
        const unsigned short* br0 = &Bs[cur][(nw + l16) * 32];
        const unsigned short* br1 = &Bs[cur][(nw + 16 + l16) * 32];
        bv0.q[0] = *(const uint4*)(br0 + 8 * half);       // B: K 8h..8h+7, 16+8h..+7
        bv0.q[1] = *(const uint4*)(br0 + 16 + 8 * half);
        bv1.q[0] = *(const uint4*)(br1 + 8 * half);
        bv1.q[1] = *(const uint4*)(br1 + 16 + 8 * half);

        acc0 = __builtin_amdgcn_wmma_f32_16x16x32_bf16(false, av.v, false, bv0.v,
                                                       (short)0, acc0, false, false);
        acc1 = __builtin_amdgcn_wmma_f32_16x16x32_bf16(false, av.v, false, bv1.v,
                                                       (short)0, acc1, false, false);
        __syncthreads();                     // done reading before overwrite
    }

    const int col0 = n0 + nw + l16;
    const int col1 = col0 + 16;
    const float b0 = bias[col0], b1 = bias[col1];
#pragma unroll
    for (int i = 0; i < 8; ++i) {
        int row = m0 + mw + i + 8 * half;
        out[(size_t)row * 512 + col0] = LRELU(acc0[i] + b0);
        out[(size_t)row * 512 + col1] = LRELU(acc1[i] + b1);
    }
}

// ===========================================================================
// K6: generic fp32 WMMA GEMM  out = lrelu(A(MxK) @ W(NxK)^T + bias), used for
// c2 (K=512,N=256) and c3 (K=256,N=128).
// ===========================================================================
__global__ void gemm_f32_wmma_kernel(const float* __restrict__ A,
                                     const float* __restrict__ W,
                                     const float* __restrict__ bias,
                                     float* __restrict__ out,
                                     int K, int N, int tilesN)
{
    const int wid  = (blockIdx.x * blockDim.x + threadIdx.x) >> 5;
    const int lane = threadIdx.x & 31;
    const int tM = wid / tilesN;
    const int tN = wid % tilesN;
    const int half = lane >> 4;
    const int l16  = lane & 15;
    const int m = tM * 16 + l16;
    const int n = tN * 16 + l16;
    const float* arow = A + (size_t)m * K;
    const float* brow = W + (size_t)n * K;
    v8f acc = {};
    for (int kk = 0; kk < K; kk += 4) {
        v2f a, b;
        a.x = arow[kk + 2 * half]; a.y = arow[kk + 2 * half + 1];
        b.x = brow[kk + 2 * half]; b.y = brow[kk + 2 * half + 1];
        acc = __builtin_amdgcn_wmma_f32_16x16x4_f32(false, a, false, b,
                                                    (short)0, acc, false, false);
    }
    float bb = bias[n];
#pragma unroll
    for (int i = 0; i < 8; ++i) {
        int row = tM * 16 + i + 8 * half;
        out[(size_t)row * N + n] = LRELU(acc[i] + bb);
    }
}

// ===========================================================================
// K7: c4 matvec (128 -> 1), no activation.
// ===========================================================================
__global__ void c4_kernel(const float* __restrict__ A, const float* __restrict__ w,
                          const float* __restrict__ bp, float* __restrict__ out)
{
    int b = blockIdx.x * blockDim.x + threadIdx.x;
    if (b < 512) {
        float s = bp[0];
        const float* ar = A + b * 128;
#pragma unroll 16
        for (int k = 0; k < 128; ++k) s += ar[k] * w[k];
        out[b] = s;
    }
}

// ===========================================================================
extern "C" void kernel_launch(void* const* d_in, const int* in_sizes, int n_in,
                              void* d_out, int out_size, void* d_ws, size_t ws_size,
                              hipStream_t stream)
{
    (void)in_sizes; (void)n_in; (void)out_size; (void)ws_size;
    const float* x     = (const float*)d_in[0];
    const float* lc1_w = (const float*)d_in[1];  const float* lc1_b = (const float*)d_in[2];
    const float* lc2_w = (const float*)d_in[3];  const float* lc2_b = (const float*)d_in[4];
    const float* lc3_w = (const float*)d_in[5];  const float* lc3_b = (const float*)d_in[6];
    const float* lc4_w = (const float*)d_in[7];  const float* lc4_b = (const float*)d_in[8];
    const float* lc5_w = (const float*)d_in[9];  const float* lc5_b = (const float*)d_in[10];
    const float* lc6_w = (const float*)d_in[11]; const float* lc6_b = (const float*)d_in[12];
    const float* ld_w  = (const float*)d_in[13]; const float* ld_b  = (const float*)d_in[14];
    const float* gc1_w = (const float*)d_in[15]; const float* gc1_b = (const float*)d_in[16];
    const float* gc2_w = (const float*)d_in[17]; const float* gc2_b = (const float*)d_in[18];
    const float* gd_w  = (const float*)d_in[19]; const float* gd_b  = (const float*)d_in[20];
    const float* w1    = (const float*)d_in[21]; const float* w2    = (const float*)d_in[22];
    const float* c1_w  = (const float*)d_in[23]; const float* c1_b  = (const float*)d_in[24];
    const float* c2_w  = (const float*)d_in[25]; const float* c2_b  = (const float*)d_in[26];
    const float* c3_w  = (const float*)d_in[27]; const float* c3_b  = (const float*)d_in[28];
    const float* c4_w  = (const float*)d_in[29]; const float* c4_b  = (const float*)d_in[30];

    char* ws = (char*)d_ws;
    size_t off = 0;
    auto alloc = [&](size_t bytes) -> void* {
        void* p = ws + off;
        off = (off + bytes + 255) & ~(size_t)255;
        return p;
    };
    float*          feat = (float*)         alloc(140800ull * 64 * 4);   // 36.0 MB
    unsigned short* U    = (unsigned short*)alloc(512ull * 17664 * 2);   // 17.3 MB
    unsigned short* W1b  = (unsigned short*)alloc(512ull * 17664 * 2);   // 17.3 MB
    float*          act1 = (float*)         alloc(512ull * 512 * 4);
    float*          act2 = (float*)         alloc(512ull * 256 * 4);
    float*          act3 = (float*)         alloc(512ull * 128 * 4);

    // c1 weights -> bf16 (9,043,968 elements; 35328 blocks exactly)
    f32_to_bf16_kernel<<<35328, 256, 0, stream>>>(c1_w, W1b, 512 * 17664);

    // local conv stack: 17600 blocks x 8 waves (one wave per sequence)
    local_branch_kernel<<<17600, 256, 0, stream>>>(x,
        lc1_w, lc1_b, lc2_w, lc2_b, lc3_w, lc3_b,
        lc4_w, lc4_b, lc5_w, lc5_b, lc6_w, lc6_b, feat);

    // ld dense via f32 WMMA: 35200 tile-waves -> 4400 blocks
    ld_gemm_kernel<<<4400, 256, 0, stream>>>(feat, ld_w, ld_b, w1, U);

    // global branch: one block per batch
    global_branch_kernel<<<512, 256, 0, stream>>>(x, gc1_w, gc1_b,
        gc2_w, gc2_b, gd_w, gd_b, w2, U);

    // c1: bf16 WMMA GEMM, 64x64 block tiles -> 8x8 grid, async LDS pipeline
    c1_gemm_bf16_kernel<<<dim3(8, 8), 256, 0, stream>>>(U, W1b, c1_b, act1);

    // c2: 512x512 @ 512x256^T -> 512 waves -> 64 blocks
    gemm_f32_wmma_kernel<<<64, 256, 0, stream>>>(act1, c2_w, c2_b, act2, 512, 256, 16);
    // c3: 512x256 @ 256x128^T -> 256 waves -> 32 blocks
    gemm_f32_wmma_kernel<<<32, 256, 0, stream>>>(act2, c3_w, c3_b, act3, 256, 128, 8);

    // c4 head
    c4_kernel<<<2, 256, 0, stream>>>(act3, c4_w, c4_b, (float*)d_out);
}
